// SceneGeometryEPModel_103079215180
// MI455X (gfx1250) — compile-verified, matrix operand order checked by reading, not run on hardware
//
#include <hip/hip_runtime.h>
#include <hip/hip_bf16.h>

typedef __attribute__((ext_vector_type(16))) _Float16 v16h;
typedef __attribute__((ext_vector_type(8)))  float    v8f;

#define HW_   (512 * 512)          // 262144 = 2^18
#define NPIX  (16 * HW_)           // 4,194,304 pixels
#define EPS_F 1e-12f

// ---------------------------------------------------------------------------
// Kernel 1: one wave32. G = W^T W via v_wmma_f32_16x16x32_f16 (8x8 padded into
// the 16x16x32 tile), power iteration for the dominant eigenvalue of G,
// sigma = sqrt(lambda). Emits the 3 per-pixel constants c[0..2] into d_ws.
// f16 precision for sigma is harmless: the final per-pixel normalization
// cancels any global scale (sigma only gates the EPS edge case).
// ---------------------------------------------------------------------------
__global__ __launch_bounds__(32) void sigma_kernel(const float* __restrict__ Wm,
                                                   float* __restrict__ cvec)
{
    __shared__ float Wsh[64];
    __shared__ float G[8][8];
    const int lane = threadIdx.x;          // 0..31
    Wsh[lane]      = Wm[lane];
    Wsh[lane + 32] = Wm[lane + 32];
    __syncthreads();

    const bool hi = lane >= 16;
    const int  mn = lane & 15;

    // A-matrix 16x32 f16: lanes 0-15 hold M=0..15, elements 0..7 are K=0..7.
    // B-matrix 32x16 f16: lanes 0-15 hold N=0..15, elements 0..7 are K=0..7.
    // Everything else (K>=8 halves, lanes 16-31 K-ranges) padded with zeros.
    v16h amat, bmat;
    for (int e = 0; e < 16; ++e) { amat[e] = (_Float16)0.f; bmat[e] = (_Float16)0.f; }
    if (!hi && mn < 8) {
        for (int e = 0; e < 8; ++e) {
            amat[e] = (_Float16)Wsh[e * 8 + mn];   // A = W^T : A[m][k] = W[k][m]
            bmat[e] = (_Float16)Wsh[e * 8 + mn];   // B = W   : B[k][n] = W[k][n]
        }
    }

    // No divergence here: WMMA requires EXEC all-ones.
    v8f acc;
    for (int e = 0; e < 8; ++e) acc[e] = 0.f;
    acc = __builtin_amdgcn_wmma_f32_16x16x32_f16(
        /*neg_a=*/false, amat, /*neg_b=*/false, bmat,
        /*c_mod=*/(short)0, acc, /*reuse_a=*/false, /*reuse_b=*/false);

    // D layout: lanes 0-15 hold column N=lane, element r is row M=r.
    if (!hi && mn < 8) {
        for (int r = 0; r < 8; ++r) G[r][mn] = acc[r];
    }
    __syncthreads();

    // Power iteration on G (8x8). Lane (lane&7) owns row r; v replicated per
    // 8-lane group so wave32 shuffles with width 8 stay coherent.
    const int r = lane & 7;
    float row[8];
    for (int j = 0; j < 8; ++j) row[j] = G[r][j];

    float v   = 0.35355339059327373f;  // 1/sqrt(8)
    float lam = 0.f;
    for (int it = 0; it < 48; ++it) {
        float u = 0.f;
        for (int j = 0; j < 8; ++j) u = fmaf(row[j], __shfl(v, j, 8), u);
        float s = u * u;
        s += __shfl_xor(s, 1, 8);
        s += __shfl_xor(s, 2, 8);
        s += __shfl_xor(s, 4, 8);
        lam = sqrtf(s);                // ||G v|| -> lambda_max(G) at convergence
        v = u / fmaxf(lam, 1e-30f);
    }

    if (lane == 0) {
        const float sigma = sqrtf(lam);                  // sigma(W) = sqrt(lambda_max(W^T W))
        const float inv   = 1.f / fmaxf(sigma, EPS_F);
        float f = 0.f;                                   // 10 Euler steps of f += DT*(1-f)
        for (int i = 0; i < 10; ++i) f += 0.1f * (1.f - f);
        cvec[0] = Wsh[1] * inv * f;                      // Wn[0][1] * f
        cvec[1] = Wsh[2] * inv * f;                      // Wn[0][2] * f
        cvec[2] = Wsh[3] * inv * f;                      // Wn[0][3] * f
    }
}

// ---------------------------------------------------------------------------
// Kernel 2: pure streaming map. 4 pixels per thread: one b128 load, three b128
// stores into the (B,3,H,W) channel planes. Bandwidth floor ~2.9 us @23.3TB/s.
// ---------------------------------------------------------------------------
__device__ __forceinline__ void norm3(float x, float c1, float c2, float c3,
                                      float& o1, float& o2, float& o3)
{
    const float nx = x * c1, ny = x * c2, nz = x * c3;
    const float nrm = sqrtf(fmaf(nx, nx, fmaf(ny, ny, nz * nz)));
    const float inv = 1.f / fmaxf(nrm, EPS_F);
    o1 = nx * inv; o2 = ny * inv; o3 = nz * inv;
}

__global__ __launch_bounds__(256) void normals_kernel(const float4* __restrict__ depth4,
                                                      const float* __restrict__ cvec,
                                                      float* __restrict__ out)
{
    const int idx = blockIdx.x * blockDim.x + threadIdx.x;   // one float4 = 4 pixels
    const float c1 = cvec[0], c2 = cvec[1], c3 = cvec[2];    // uniform -> scalar loads

    const int p  = idx << 2;
    const int bb = p >> 18;            // p / HW_
    const int hw = p & (HW_ - 1);      // p % HW_

    const float4 x = depth4[idx];
    float4 o1, o2, o3;
    norm3(x.x, c1, c2, c3, o1.x, o2.x, o3.x);
    norm3(x.y, c1, c2, c3, o1.y, o2.y, o3.y);
    norm3(x.z, c1, c2, c3, o1.z, o2.z, o3.z);
    norm3(x.w, c1, c2, c3, o1.w, o2.w, o3.w);

    const size_t base = (size_t)bb * (3u * HW_) + (size_t)hw;
    *(float4*)(out + base)            = o1;   // channel 0 plane
    *(float4*)(out + base + HW_)      = o2;   // channel 1 plane
    *(float4*)(out + base + 2 * HW_)  = o3;   // channel 2 plane
}

extern "C" void kernel_launch(void* const* d_in, const int* in_sizes, int n_in,
                              void* d_out, int out_size, void* d_ws, size_t ws_size,
                              hipStream_t stream)
{
    const float* depth = (const float*)d_in[0];   // (16,1,512,512) f32
    const float* W     = (const float*)d_in[1];   // (8,8) f32
    float*       cvec  = (float*)d_ws;            // 3 floats of scratch
    float*       out   = (float*)d_out;           // (16,3,512,512) f32

    sigma_kernel<<<1, 32, 0, stream>>>(W, cvec);

    const int threads = 256;
    const int blocks  = (NPIX / 4) / threads;     // 4096 blocks, 4 pixels/thread
    normals_kernel<<<blocks, threads, 0, stream>>>((const float4*)depth, cvec, out);
}
